// LeNet_MNIST_86303072846337
// MI455X (gfx1250) — matrix operand order repaired and checked by reading, not compiled, not run on hardware
//
#include <hip/hip_runtime.h>
#include <hip/hip_bf16.h>

// ---------------------------------------------------------------------------
// Spiking LeNet, T=512 timesteps, B=32. Persistent kernel: 1 block = 1 sample,
// whole network state + f16 weights resident in CDNA5's 320KB-per-WG LDS.
// conv1/conv2 via v_wmma_f32_16x16x32_f16 (spikes are {0,1} -> f16 exact).
// B-fragment gathers are UNCONDITIONAL: padded LDS buffers keep addresses
// in-bounds; zero A-weights / discarded columns annihilate the padding reads.
// fc1 weights arrive via the Tensor Data Mover (tensor_load_to_lds +
// s_wait_tensorcnt) into an LDS staging region, then get converted to f16.
// ---------------------------------------------------------------------------

typedef __attribute__((ext_vector_type(16))) _Float16 v16h;
typedef __attribute__((ext_vector_type(8)))  _Float16 v8h;
typedef __attribute__((ext_vector_type(8)))  float    v8f;
typedef __attribute__((ext_vector_type(4)))  unsigned v4u;
typedef __attribute__((ext_vector_type(4)))  int      v4i;
typedef __attribute__((ext_vector_type(8)))  int      v8i;

#define T_STEPS 512
#define BATCH   32
#define RATE_DT 0.1953125f   /* MAX_RATE * DELTA_T = 200 * 2^-10 */

struct Smem {
    // --- input spikes (current timestep), padded to 34x32 rows, f16 {0,1} ---
    alignas(32) _Float16 xs[1088];        // rows 32..33 are zero padding
    // --- weights, pre-swizzled / converted at kernel start ---
    alignas(32) _Float16 w1A[512];        // conv1 A-frag [lane][slot], M=6 K=25 zero-padded to 16x32
    alignas(32) _Float16 w2A[2560];       // conv2 A-frags: 5 K-chunks of [lane][slot], K>=150 zeroed
    alignas(32) _Float16 w3h[48000];      // conv3 120x400 row-major f16 (row = 800B, 32B-mult)
    alignas(16) _Float16 fw1h[10080];     // fc1 84x120 f16 (row = 240B, 16B-mult)
    float    fw2f[840];                   // fc2 10x84 f32
    // --- LIF states (zeroed every launch). m*a = mem1, m*b = mem2 ---
    // NOTE: m1a..sp1 (47KB contiguous) doubles as TDM staging for fw1 (40320B)
    // during setup, before the state-zeroing pass.
    float    m1a[4704], m1b[4704];  _Float16 sp1[4704];   // 6x28x28
    float    m2a[1176], m2b[1176];  _Float16 sp2[1400];   // 6x14x14 (+224 pad halves, zeroed)
    float    m3a[1600], m3b[1600];  _Float16 sp3[1600];   // 16x10x10
    float    m4a[400],  m4b[400];
    alignas(32) _Float16 sp4[400];                        // 16x5x5
    float    m5a[120],  m5b[120];
    alignas(16) _Float16 sp5h[120];                       // 120
    float    m6a[84],   m6b[84];    _Float16 sp6h[84];    // 84
    float    m7a[10],   m7b[10];    float    sp7[10];     // 10
    float    acc[12];                                     // 10 used (+pad)
};

extern __shared__ char smem_raw[];

// stateless hash -> uniform [0,1)
__device__ __forceinline__ float rng01(unsigned v) {
    v ^= v >> 16; v *= 0x7feb352dU;
    v ^= v >> 15; v *= 0x846ca68bU;
    v ^= v >> 16;
    return (float)(v >> 8) * (1.0f / 16777216.0f);
}

// two-compartment LIF with hard reset (spike zeroes decayed state); returns new spike, updates m1/m2
__device__ __forceinline__ float lif(float drive, float& m1, float& m2, float spike_prev) {
    m1 = m1 * 0.25f * (1.0f - spike_prev) + drive;
    m2 = m2 * 0.25f * (1.0f - spike_prev) + 0.5f * m1;
    return (m2 > 0.3f) ? 1.0f : 0.0f;
}

// 16-bit A-matrix (16x32) K index for (lane quadrant q, slot i): contiguous K = 16*q + i (matches B)
__device__ __forceinline__ int klutA(int q, int i) {
    return 16 * q + i;
}

#define HAVE_TDM __has_builtin(__builtin_amdgcn_tensor_load_to_lds)

__global__ void __launch_bounds__(256)
snn_lenet_kernel(const float* __restrict__ image, const float* __restrict__ w1,
                 const float* __restrict__ w2, const float* __restrict__ w3,
                 const float* __restrict__ fw1, const float* __restrict__ fw2,
                 float* __restrict__ out)
{
    Smem* S = reinterpret_cast<Smem*>(smem_raw);
    const int tid  = threadIdx.x;
    const int lane = tid & 31;
    const int wave = tid >> 5;
    const int b    = blockIdx.x;

    // Warm the weight path in L2 (global_prefetch_b8)
    __builtin_prefetch(w3  + (tid << 5), 0, 0);
    __builtin_prefetch(image + b * 1024 + (tid << 2), 0, 0);

#if HAVE_TDM
    // ---- TDM: DMA fc1 weights (84x120 f32 = 40320B) global -> LDS staging ----
    // D# per CDNA5 ISA ch.8: 1D tile of 10080 4-byte elements, no multicast.
    if (wave == 0) {
        unsigned long long ga = (unsigned long long)(const void*)fw1;
        unsigned ldsoff = (unsigned)offsetof(Smem, m1a);   // dyn-LDS base == 0
        v4u g0 = { 1u,                                      // count=1, user desc
                   ldsoff,                                   // lds_addr [63:32]
                   (unsigned)ga,                             // global_addr lo
                   (unsigned)((ga >> 32) & 0x1FFFFFFu) | 0x80000000u }; // hi | type=2
        v8i g1 = { 0x20000,                 // wg_mask=0, data_size=2 (4B)
                   (int)(10080u << 16),     // tensor_dim0[15:0] << 16 (=10080)
                   0x10000,                 // tensor_dim0[31:16]=0, tensor_dim1=1
                   (int)(10080u << 16),     // tile_dim0 = 10080
                   1,                       // tile_dim1 = 1, tile_dim2 = 0
                   10080,                   // tensor_dim0_stride lo
                   0, 0 };                  // stride hi, dim1_stride
        v4i gz = { 0, 0, 0, 0 };
#if defined(__clang_major__) && (__clang_major__ >= 23)
        v8i gz8 = { 0, 0, 0, 0, 0, 0, 0, 0 };
        __builtin_amdgcn_tensor_load_to_lds(g0, g1, gz, gz, gz8, 0);
#else
        __builtin_amdgcn_tensor_load_to_lds(g0, g1, gz, gz, 0);
#endif
    }
#endif

    // ---- one-time setup: swizzle/convert weights into LDS (overlaps DMA) ----
    for (int e = tid; e < 1088; e += 256) S->xs[e] = (_Float16)0;   // incl. pad rows
    for (int e = tid; e < 512; e += 256) {          // conv1 A-fragment
        int l = e >> 4, i = e & 15, q = l >> 4, M = l & 15;
        int K = klutA(q, i);
        S->w1A[e] = (M < 6 && K < 25) ? (_Float16)w1[M * 25 + K] : (_Float16)0;
    }
    for (int e = tid; e < 2560; e += 256) {         // conv2 A-fragments (5 K-chunks)
        int ch = e >> 9, r = e & 511, l = r >> 4, i = r & 15, q = l >> 4, M = l & 15;
        int K = ch * 32 + klutA(q, i);
        S->w2A[e] = (K < 150) ? (_Float16)w2[M * 150 + K] : (_Float16)0;
    }
    for (int e = tid; e < 48000; e += 256) S->w3h[e]  = (_Float16)w3[e];
    for (int e = tid; e < 840;   e += 256) S->fw2f[e] = fw2[e];

#if HAVE_TDM
    if (wave == 0) __builtin_amdgcn_s_wait_tensorcnt(0);   // DMA -> LDS complete
    __syncthreads();
    {   // convert staged f32 fc1 weights (in m1a..) to f16
        const float* stage = (const float*)S->m1a;
        for (int e = tid; e < 10080; e += 256) S->fw1h[e] = (_Float16)stage[e];
    }
    __syncthreads();
#else
    for (int e = tid; e < 10080; e += 256) S->fw1h[e] = (_Float16)fw1[e];
#endif

    {   // zero all membrane/spike state (incl. sp2 pad) + acc: contiguous tail
        unsigned* z0 = (unsigned*)S->m1a;
        unsigned* z1 = (unsigned*)(S->acc + 12);
        for (unsigned* p = z0 + tid; p < z1; p += 256) *p = 0u;
    }
    __syncthreads();

    const float* img = image + b * 1024;

    for (int t = 0; t < T_STEPS; ++t) {
        // ---- stage 1: Bernoulli input spikes (rows 0..31 only) ----
        for (int i = tid; i < 1024; i += 256) {
            unsigned s = 0x9E3779B9u * (unsigned)(t + 1) ^ 0x85EBCA6Bu * (unsigned)(b * 1024 + i + 1);
            S->xs[i] = (img[i] * RATE_DT > rng01(s)) ? (_Float16)1 : (_Float16)0;
        }
        __syncthreads();

        // ---- stage 2: conv1 (im2col GEMM 16x32 x 32x16 WMMA) + LIF1 ----
        {
            v16h a1 = *(const v16h*)&S->w1A[lane * 16];
            int q = lane >> 4, col = lane & 15;
            for (int tile = wave; tile < 49; tile += 8) {    // N = 784 = 49*16
                int n = tile * 16 + col;
                int oy = n / 28, ox = n % 28;
                int base = oy * 32 + ox;
                v16h bf;
                #pragma unroll
                for (int i = 0; i < 16; ++i) {
                    int K = q * 16 + i;                       // B layout: K = 16*quad + slot
                    bf[i] = S->xs[base + (K / 5) * 32 + (K % 5)];  // unconditional, padded
                }
                v8f c = {};
                c = __builtin_amdgcn_wmma_f32_16x16x32_f16(false, a1, false, bf,
                                                           (short)0, c, false, false);
                #pragma unroll
                for (int j = 0; j < 8; ++j) {                 // D: M = j + 8*quad, N = n
                    int M = j + 8 * q;
                    if (M < 6) {
                        int idx = M * 784 + n;
                        float m1 = S->m1a[idx], m2 = S->m1b[idx];
                        float ns = lif(c[j], m1, m2, (float)S->sp1[idx]);
                        S->m1a[idx] = m1; S->m1b[idx] = m2; S->sp1[idx] = (_Float16)ns;
                    }
                }
            }
        }
        __syncthreads();

        // ---- stage 3: avgpool 2x2 + LIF2 (6x14x14) ----
        for (int i = tid; i < 1176; i += 256) {
            int c = i / 196, r = i % 196, y = r / 14, x = r % 14;
            int bix = c * 784 + (2 * y) * 28 + 2 * x;
            float drive = 0.25f * ((float)S->sp1[bix] + (float)S->sp1[bix + 1] +
                                   (float)S->sp1[bix + 28] + (float)S->sp1[bix + 29]);
            float m1 = S->m2a[i], m2 = S->m2b[i];
            float ns = lif(drive, m1, m2, (float)S->sp2[i]);
            S->m2a[i] = m1; S->m2b[i] = m2; S->sp2[i] = (_Float16)ns;
        }
        __syncthreads();

        // ---- stage 4: conv2 (M=16, K=150->5x32, N=100->7x16 WMMA) + LIF3 ----
        {
            int q = lane >> 4, col = lane & 15;
            for (int tile = wave; tile < 7; tile += 8) {      // waves 0..6
                int n = tile * 16 + col;                       // 0..111; valid < 100
                int oy = n / 10, ox = n % 10;                  // oy<=11: stays in padded sp2
                int base = oy * 14 + ox;
                v8f c = {};
                #pragma unroll
                for (int ch = 0; ch < 5; ++ch) {
                    v16h a = *(const v16h*)&S->w2A[ch * 512 + lane * 16];
                    v16h bf;
                    #pragma unroll
                    for (int i = 0; i < 16; ++i) {
                        int K = ch * 32 + q * 16 + i;          // < 160, cc<=6 in padded sp2
                        int cc = K / 25, r = K % 25;
                        bf[i] = S->sp2[cc * 196 + base + (r / 5) * 14 + (r % 5)];
                    }
                    c = __builtin_amdgcn_wmma_f32_16x16x32_f16(false, a, false, bf,
                                                               (short)0, c, false, false);
                }
                bool nv = (n < 100);
                #pragma unroll
                for (int j = 0; j < 8; ++j) {
                    int M = j + 8 * q;
                    if (nv) {
                        int idx = M * 100 + n;
                        float m1 = S->m3a[idx], m2 = S->m3b[idx];
                        float ns = lif(c[j], m1, m2, (float)S->sp3[idx]);
                        S->m3a[idx] = m1; S->m3b[idx] = m2; S->sp3[idx] = (_Float16)ns;
                    }
                }
            }
        }
        __syncthreads();

        // ---- stage 5: avgpool 2x2 + LIF4 (16x5x5) ----
        for (int i = tid; i < 400; i += 256) {
            int c = i / 25, r = i % 25, y = r / 5, x = r % 5;
            int bix = c * 100 + (2 * y) * 10 + 2 * x;
            float drive = 0.25f * ((float)S->sp3[bix] + (float)S->sp3[bix + 1] +
                                   (float)S->sp3[bix + 10] + (float)S->sp3[bix + 11]);
            float m1 = S->m4a[i], m2 = S->m4b[i];
            float ns = lif(drive, m1, m2, (float)S->sp4[i]);
            S->m4a[i] = m1; S->m4b[i] = m2; S->sp4[i] = (_Float16)ns;
        }
        __syncthreads();

        // ---- stage 6: conv3 as 120x400 matvec + LIF5 (vector v8h LDS loads) ----
        if (tid < 120) {
            const v8h* wr = (const v8h*)&S->w3h[tid * 400];
            const v8h* sv = (const v8h*)S->sp4;
            float d = 0.0f;
            #pragma unroll 5
            for (int k = 0; k < 50; ++k) {
                v8h a = wr[k], s = sv[k];
                #pragma unroll
                for (int j = 0; j < 8; ++j) d += (float)a[j] * (float)s[j];
            }
            float m1 = S->m5a[tid], m2 = S->m5b[tid];
            float ns = lif(d, m1, m2, (float)S->sp5h[tid]);
            S->m5a[tid] = m1; S->m5b[tid] = m2; S->sp5h[tid] = (_Float16)ns;
        }
        __syncthreads();

        // ---- stage 7: fc1 84x120 + LIF6 (vector v8h LDS loads) ----
        if (tid < 84) {
            const v8h* wr = (const v8h*)&S->fw1h[tid * 120];
            const v8h* sv = (const v8h*)S->sp5h;
            float d = 0.0f;
            #pragma unroll 5
            for (int k = 0; k < 15; ++k) {
                v8h a = wr[k], s = sv[k];
                #pragma unroll
                for (int j = 0; j < 8; ++j) d += (float)a[j] * (float)s[j];
            }
            float m1 = S->m6a[tid], m2 = S->m6b[tid];
            float ns = lif(d, m1, m2, (float)S->sp6h[tid]);
            S->m6a[tid] = m1; S->m6b[tid] = m2; S->sp6h[tid] = (_Float16)ns;
        }
        __syncthreads();

        // ---- stage 8: fc2 10x84 + LIF7 + spike accumulation ----
        if (tid < 10) {
            float d = 0.0f;
            const float* wr = &S->fw2f[tid * 84];
            for (int k = 0; k < 84; ++k) d += wr[k] * (float)S->sp6h[k];
            float m1 = S->m7a[tid], m2 = S->m7b[tid];
            float ns = lif(d, m1, m2, S->sp7[tid]);
            S->m7a[tid] = m1; S->m7b[tid] = m2; S->sp7[tid] = ns;
            S->acc[tid] += ns;
        }
        __syncthreads();   // protect xs/sp6 before next timestep overwrites
    }

    if (tid < 10) out[b * 10 + tid] = S->acc[tid] * (1.0f / (float)T_STEPS);
}

extern "C" void kernel_launch(void* const* d_in, const int* in_sizes, int n_in,
                              void* d_out, int out_size, void* d_ws, size_t ws_size,
                              hipStream_t stream) {
    (void)in_sizes; (void)n_in; (void)out_size; (void)d_ws; (void)ws_size;
    const float* image = (const float*)d_in[0];
    const float* w1    = (const float*)d_in[1];
    const float* w2    = (const float*)d_in[2];
    const float* w3    = (const float*)d_in[3];
    const float* fw1   = (const float*)d_in[4];
    const float* fw2   = (const float*)d_in[5];
    float* out = (float*)d_out;

    // Opt in to >64KB dynamic LDS (CDNA5: 320KB per workgroup). Idempotent.
    (void)hipFuncSetAttribute((const void*)snn_lenet_kernel,
                              hipFuncAttributeMaxDynamicSharedMemorySize,
                              (int)sizeof(Smem));

    snn_lenet_kernel<<<BATCH, 256, sizeof(Smem), stream>>>(image, w1, w2, w3, fw1, fw2, out);
}
